// LossTriplet_72980084293769
// MI455X (gfx1250) — compile-verified, matrix operand order checked by reading, not run on hardware
//
#include <hip/hip_runtime.h>
#include <hip/hip_bf16.h>
#include <math.h>

// ---------------------------------------------------------------------------
// Problem constants (match the JAX reference)
// ---------------------------------------------------------------------------
#define NUM_CLASSES 5
#define KMAX        5
#define DIM         32                       // PROTO_DIM
#define NVOX        (4*64*64*64)             // 1,048,576 voxels
#define VOX_PER_B   (64*64*64)               // 262,144
#define TILES       (NVOX/16)                // 65,536 WMMA voxel tiles
#define SLOTS       4                        // len(labeled_batch)
#define BUCKETS     (SLOTS*KMAX)             // 20 (class-slot, k) centroids
#define CK          (NUM_CLASSES*KMAX)       // 25 prototypes
#define MAX_ANCHORS 2048
#define ANCH_STRIDE (NVOX/MAX_ANCHORS)       // 512
#define KM_ITERS    5
#define EPSF        1e-12f
#define MARGINF     1.0f

// assign-kernel geometry: 256 wgs x 256 thr = 2048 wave32, 32 tiles per wave
#define ASSIGN_WGS     256
#define ASSIGN_THREADS 256
#define ASSIGN_WAVES   (ASSIGN_WGS*ASSIGN_THREADS/32)   // 2048
#define WAVES_PER_WG   (ASSIGN_THREADS/32)              // 8
#define PARTCOLS       (BUCKETS*33)                     // 20 x (32 dims + count)
#define TILE_BYTES     (16*DIM*2)                       // 1024 B per voxel tile

#define TRI_WAVES      (MAX_ANCHORS/16)                 // 128

typedef _Float16 v16h __attribute__((ext_vector_type(16)));
typedef _Float16 v8h  __attribute__((ext_vector_type(8)));
typedef float    v8f  __attribute__((ext_vector_type(8)));

// ---------------------------------------------------------------------------
// Workspace layout (bytes).  Total ~72.6 MB.
// ---------------------------------------------------------------------------
static constexpr size_t OFF_A      = 0;                           // fp16 [N][32] normalized feats
static constexpr size_t SZ_A       = (size_t)NVOX * DIM * 2;      // 67,108,864
static constexpr size_t OFF_PART   = OFF_A + SZ_A;                // f32 [2048 waves][20][33]
static constexpr size_t SZ_PART    = (size_t)ASSIGN_WAVES * PARTCOLS * 4;
static constexpr size_t OFF_SUMS   = OFF_PART + SZ_PART;          // f32 [20][33] reduced
static constexpr size_t OFF_CENT   = OFF_SUMS   + 2688;           // f32 [20][32] centroids
static constexpr size_t OFF_CC2    = OFF_CENT   + 2560;           // f32 [20] |cent|^2
static constexpr size_t OFF_CENTA  = OFF_CC2    + 128;            // fp16 [32][32] centroid A-tiles (16B aligned)
static constexpr size_t OFF_INITPT = OFF_CENTA  + 2048;           // f32 [20][32] init/fallback points
static constexpr size_t OFF_PROTO  = OFF_INITPT + 2560;           // f32 [25][32] normalized prototypes
static constexpr size_t OFF_PROTOA = OFF_PROTO  + 3200;           // fp16 [32][32] proto A-tiles
static constexpr size_t OFF_CLS    = OFF_PROTOA + 2048;           // f32 [4][33] class sum+count
static constexpr size_t OFF_CFIN   = OFF_CLS    + 544;            // f32 [20][32] final centroids
static constexpr size_t OFF_TRI    = OFF_CFIN   + 2560;           // f32 [128][2] triplet partials
static constexpr size_t OFF_SCAL   = OFF_TRI    + 1024;           // f32 scalars: [0]=anchor [1]=compact

__device__ __forceinline__ unsigned hashu(unsigned x) {
  x ^= x >> 16; x *= 0x7feb352du; x ^= x >> 15; x *= 0x846ca68bu; x ^= x >> 16;
  return x;
}

// Load one 16x32 fp16 operand tile (A or B role) for V_WMMA_F32_16X16X32_F16.
// Per ISA 7.12.2 (16-bit A 16x32): lanes 0-15 hold line (lane&15) with
// K in {0..7, 16..23}; lanes 16-31 hold K in {8..15, 24..31}.  `row` points
// to this lane's 32-element fp16 row; rows are 64B aligned so both 16B
// chunks are 16B aligned.  Works for global or LDS-staged rows (address
// space is recovered by the compiler after inlining).
__device__ __forceinline__ v16h load_row_tile(int lane, const _Float16* __restrict__ row) {
  const int kb = (lane < 16) ? 0 : 8;
  v8h lo = *(const v8h*)(row + kb);
  v8h hi = *(const v8h*)(row + kb + 16);
  v16h r;
#pragma unroll
  for (int j = 0; j < 8; ++j) { r[j] = lo[j]; r[j + 8] = hi[j]; }
  return r;
}

// ---------------------------------------------------------------------------
// Pass 1: transpose (B,D,Z,Y,X)->(N,D), L2-normalize rows, store fp16.
// One HBM read of feats (134 MB) + one 64 MB write; everything downstream
// stays resident in the 192 MB L2.
// ---------------------------------------------------------------------------
__global__ void __launch_bounds__(256) k_normalize(const float* __restrict__ feats,
                                                   _Float16* __restrict__ A) {
  const int n = blockIdx.x * blockDim.x + threadIdx.x;   // grid == NVOX exactly
  const int b = n >> 18;                                  // / VOX_PER_B
  const int s = n & (VOX_PER_B - 1);
  const float* base = feats + (size_t)b * DIM * VOX_PER_B + s;
  float v[DIM];
  float ss = 0.f;
#pragma unroll
  for (int d = 0; d < DIM; ++d) {
    float x = base[(size_t)d * VOX_PER_B];
    v[d] = x; ss += x * x;
  }
  const float inv = rsqrtf(ss + EPSF);
  _Float16* row = A + (size_t)n * DIM;
#pragma unroll
  for (int d = 0; d < DIM; ++d) row[d] = (_Float16)(v[d] * inv);
}

// ---------------------------------------------------------------------------
// Normalize prototypes; emit f32 copy and fp16 zero-padded 32x32 A-tile buf.
// ---------------------------------------------------------------------------
__global__ void k_protopack(const float* __restrict__ protos,
                            float* __restrict__ pn, _Float16* __restrict__ pa) {
  const int r = threadIdx.x;                 // 0..31
  if (r < CK) {
    const float* p = protos + r * DIM;
    float ss = 0.f;
#pragma unroll
    for (int d = 0; d < DIM; ++d) ss += p[d] * p[d];
    const float inv = rsqrtf(ss + EPSF);
#pragma unroll
    for (int d = 0; d < DIM; ++d) {
      const float x = p[d] * inv;
      pn[r * DIM + d] = x;
      pa[r * DIM + d] = (_Float16)x;
    }
  } else {
#pragma unroll
    for (int d = 0; d < DIM; ++d) pa[r * DIM + d] = (_Float16)0.f;
  }
}

// ---------------------------------------------------------------------------
// Deterministic k-means init: thread (slot,k) stride-walks the gt mask with a
// full-cycle odd increment mod 2^20 until it hits a voxel of its class.
// (Substitute for the reference's gumbel-top-k random init; deterministic.)
// ---------------------------------------------------------------------------
__global__ void k_initcent(const int* __restrict__ gt, const int* __restrict__ lab,
                           const _Float16* __restrict__ A,
                           float* __restrict__ cent, float* __restrict__ initpt,
                           float* __restrict__ cc2, _Float16* __restrict__ centA) {
  const int t = threadIdx.x;                 // 0..31
  if (t >= BUCKETS) {                        // zero-pad centroid A-tile rows 20..31
#pragma unroll
    for (int d = 0; d < DIM; ++d) centA[t * DIM + d] = (_Float16)0.f;
    return;
  }
  const int slot = t / KMAX;
  const int c = lab[slot];
  unsigned n = (unsigned)((t * 402653189u + 12345u) & (NVOX - 1));
  int found = -1;
  for (int step = 0; step < NVOX; ++step) {
    if (gt[n] == c) { found = (int)n; break; }
    n = (n + 2654435769u) & (NVOX - 1);      // odd increment -> full cycle
  }
  float ss = 0.f;
#pragma unroll
  for (int d = 0; d < DIM; ++d) {
    const float x = (found >= 0) ? (float)A[(size_t)found * DIM + d] : 0.f;
    cent[t * DIM + d] = x;
    initpt[t * DIM + d] = x;
    centA[t * DIM + d] = (_Float16)x;
    ss += x * x;
  }
  cc2[t] = ss;
}

// ---------------------------------------------------------------------------
// K-means assignment pass (the hot kernel).
//   * A-matrix = 20 centroids (2 tiles, loop-invariant in VGPRs)
//   * B-matrix = 16 voxels, staged through LDS with the gfx1250 async copy
//     engine: each tile is 16 consecutive 64B rows = one contiguous 1 KB
//     block -> two GLOBAL_LOAD_ASYNC_TO_LDS_B128 per tile, double-buffered
//     and pipelined on ASYNCcnt.  The staged tile feeds both the WMMA
//     B-operand and the per-dim accumulation reads (ds_load instead of vmem).
//   * Per-class argmin is computed split across the lane pair (each half
//     scans only its resident similarities), then merged with a single
//     (best,idx) exchange -- 2 shuffles/tile instead of 12.
//   * Deterministic: per-wave partial sums, no float atomics.
// ---------------------------------------------------------------------------
__global__ void __launch_bounds__(ASSIGN_THREADS) k_assign(
    const _Float16* __restrict__ A, const _Float16* __restrict__ centA,
    const float* __restrict__ cc2, const int* __restrict__ gt,
    const int* __restrict__ lab, float* __restrict__ partial) {
  __shared__ __attribute__((aligned(16))) _Float16 stage[WAVES_PER_WG][2][16 * DIM]; // 16 KB
  const int lane = threadIdx.x & 31;
  const int wid  = threadIdx.x >> 5;
  const int wave = (blockIdx.x * blockDim.x + threadIdx.x) >> 5;
  const int m = lane & 15;
  const int lab0 = lab[0], lab1 = lab[1], lab2 = lab[2], lab3 = lab[3];

  // Loop-invariant centroid tiles (rows 0..15 and 16..31, rows>=20 are zero).
  const v16h a0 = load_row_tile(lane, centA + m * DIM);
  const v16h a1 = load_row_tile(lane, centA + (16 + m) * DIM);

  float c2[BUCKETS];
#pragma unroll
  for (int i = 0; i < BUCKETS; ++i) c2[i] = cc2[i];

  float accS[BUCKETS], accC[BUCKETS];
#pragma unroll
  for (int i = 0; i < BUCKETS; ++i) { accS[i] = 0.f; accC[i] = 0.f; }

  // LDS byte offsets of this wave's two staging buffers (low 32 bits of the
  // generic LDS-aperture address are the LDS offset).
  const unsigned buf[2] = {
      (unsigned)(uintptr_t)&stage[wid][0][0],
      (unsigned)(uintptr_t)&stage[wid][1][0]};
  const unsigned long gA = (unsigned long)(uintptr_t)A;

  // async-copy one contiguous 1 KB voxel tile into an LDS buffer
  auto issue_tile = [&](int t, unsigned bufoff) {
    const unsigned long g = gA + (unsigned long)t * TILE_BYTES + (unsigned)(lane * 16);
    const unsigned d = bufoff + (unsigned)(lane * 16);
    asm volatile("global_load_async_to_lds_b128 %0, %1, off" :: "v"(d), "v"(g) : "memory");
    asm volatile("global_load_async_to_lds_b128 %0, %1, off"
                 :: "v"(d + 512u), "v"(g + 512ul) : "memory");
  };

  int cur = 0;
  issue_tile(wave, buf[0]);                       // prime the pipeline

  for (int t = wave; t < TILES; t += ASSIGN_WAVES) {
    const int tn = t + ASSIGN_WAVES;
    if (tn < TILES) {                             // wave-uniform branch
      // previous reads of buf[cur^1] must have landed before the async
      // engine may overwrite it (WAR), then keep 2 loads in flight.
      asm volatile("s_wait_dscnt 0x0" ::: "memory");
      issue_tile(tn, buf[cur ^ 1]);
      asm volatile("s_wait_asynccnt 0x2" ::: "memory");  // current tile done
    } else {
      asm volatile("s_wait_asynccnt 0x0" ::: "memory");
    }
    __builtin_amdgcn_fence(__ATOMIC_ACQUIRE, "workgroup");  // LDS visibility

    const _Float16* tp = &stage[wid][cur][0];
    const v16h b = load_row_tile(lane, tp + m * DIM);

    v8f d1 = {}, d2 = {};
    d1 = __builtin_amdgcn_wmma_f32_16x16x32_f16(false, a0, false, b, (short)0, d1, false, false);
    d2 = __builtin_amdgcn_wmma_f32_16x16x32_f16(false, a1, false, b, (short)0, d2, false, false);

    const int base = t * 16;
    const int g = gt[base + m];
    int slot = (g == lab0) ? 0 : (g == lab1) ? 1 : (g == lab2) ? 2 : (g == lab3) ? 3 : -1;
    if (g <= 0 || g >= NUM_CLASSES) slot = -1;
    const int r0 = slot * KMAX;

    // split argmin: each half-lane scans only the sims it holds
    // (lo: cents 0..7 in d1, 16..19 in d2; hi: cents 8..15 in d1)
    float best = 1e30f; int bi = -1;
#pragma unroll
    for (int idx = 0; idx < BUCKETS; ++idx) {
      float sv; bool cov;
      if (idx < 8)       { sv = d1[idx];      cov = (lane < 16); }
      else if (idx < 16) { sv = d1[idx - 8];  cov = (lane >= 16); }
      else               { sv = d2[idx - 16]; cov = (lane < 16); }
      const bool inr = cov && (slot >= 0) && (idx >= r0) && (idx < r0 + KMAX);
      const float dd = c2[idx] - 2.f * sv;     // d^2 = |c|^2 - 2 s + 1 (const)
      if (inr && dd < best) { best = dd; bi = idx; }
    }
    const float obest = __shfl_xor(best, 16, 32);
    const int   obi   = __shfl_xor(bi,   16, 32);
    int bkt;                                      // deterministic tie-break
    if (bi < 0)            bkt = obi;
    else if (obi < 0)      bkt = bi;
    else if (obest < best) bkt = obi;
    else if (best < obest) bkt = bi;
    else                   bkt = (bi < obi) ? bi : obi;

    // accumulate from the LDS-staged tile: lane == feature dim
#pragma unroll 4
    for (int v = 0; v < 16; ++v) {
      const int bv = __shfl(bkt, v, 32);          // wave-uniform broadcast
      if (bv >= 0) {
        const float val = (float)tp[v * DIM + lane];
#pragma unroll
        for (int q = 0; q < BUCKETS; ++q) accS[q] += (q == bv) ? val : 0.f;
        if (lane == 0) {
#pragma unroll
          for (int q = 0; q < BUCKETS; ++q) accC[q] += (q == bv) ? 1.f : 0.f;
        }
      }
    }
    cur ^= 1;
  }

  float* p = partial + (size_t)wave * PARTCOLS;
#pragma unroll
  for (int q = 0; q < BUCKETS; ++q) {
    p[q * 33 + lane] = accS[q];
    if (lane == 0) p[q * 33 + 32] = accC[q];
  }
}

// Ordered (deterministic) reduction of per-wave partials.
__global__ void k_reduce(const float* __restrict__ partial, float* __restrict__ sums) {
  const int i = threadIdx.x;
  if (i >= PARTCOLS) return;
  float acc = 0.f;
  for (int w = 0; w < ASSIGN_WAVES; ++w) acc += partial[(size_t)w * PARTCOLS + i];
  sums[i] = acc;
}

// Centroid update: newc = sum/cnt, empty clusters fall back to the init point
// (deterministic substitute for the reference's random fallback).  One wave
// per bucket; lane==dim; wave-reduce |c|^2.  Iter 0 also captures per-class
// count+sum (= sum over this slot's buckets) for the cnt<K mean_rep branch.
__global__ void k_update(const float* __restrict__ sums, const float* __restrict__ initpt,
                         float* __restrict__ cent, float* __restrict__ cc2,
                         _Float16* __restrict__ centA, float* __restrict__ clsstats,
                         int iter) {
  const int bkt = threadIdx.x >> 5;          // launch 640 threads = 20 waves
  const int lane = threadIdx.x & 31;
  if (bkt >= BUCKETS) return;
  const float cnt = sums[bkt * 33 + 32];
  const float sv = sums[bkt * 33 + lane];
  const float nv = (cnt > 0.f) ? (sv / cnt) : initpt[bkt * DIM + lane];
  cent[bkt * DIM + lane] = nv;
  centA[bkt * DIM + lane] = (_Float16)nv;
  float sq = nv * nv;
#pragma unroll
  for (int off = 16; off; off >>= 1) sq += __shfl_xor(sq, off, 32);
  if (lane == 0) cc2[bkt] = sq;
  if (iter == 0 && (bkt % KMAX) == 0) {
    const int slot = bkt / KMAX;
    float cs = 0.f, cc = 0.f;
    for (int kk = 0; kk < KMAX; ++kk) {
      cs += sums[(slot * KMAX + kk) * 33 + lane];
      cc += sums[(slot * KMAX + kk) * 33 + 32];
    }
    clsstats[slot * 33 + lane] = cs;
    if (lane == 0) clsstats[slot * 33 + 32] = cc;
  }
}

// cond(cnt>=K, kmeans, mean_rep) + final normalize.
__global__ void k_finalize(const float* __restrict__ cent, const float* __restrict__ clsstats,
                           float* __restrict__ cfin) {
  const int t = threadIdx.x;
  if (t >= BUCKETS) return;
  const int slot = t / KMAX;
  const float ccnt = clsstats[slot * 33 + 32];
  float row[DIM], ss = 0.f;
#pragma unroll
  for (int d = 0; d < DIM; ++d) {
    const float x = (ccnt >= (float)KMAX) ? cent[t * DIM + d]
                                          : clsstats[slot * 33 + d] / fmaxf(ccnt, 1.f);
    row[d] = x; ss += x * x;
  }
  const float inv = rsqrtf(ss + EPSF);
#pragma unroll
  for (int d = 0; d < DIM; ++d) cfin[t * DIM + d] = row[d] * inv;
}

__global__ void k_anchor(const float* __restrict__ pn, const float* __restrict__ cfin,
                         const float* __restrict__ clsstats, const int* __restrict__ lab,
                         float* __restrict__ scal) {
  if (threadIdx.x != 0) return;
  float al = 0.f, ac = 0.f;
  for (int slot = 0; slot < SLOTS; ++slot) {
    const int c = lab[slot];
    int cs = c < 0 ? 0 : (c > NUM_CLASSES - 1 ? NUM_CLASSES - 1 : c);
    const float cnt = clsstats[slot * 33 + 32];
    const float valid = (c != 0 && c < NUM_CLASSES && cnt > 0.f) ? 1.f : 0.f;
    float l = 0.f;
    for (int k = 0; k < KMAX; ++k)
      for (int d = 0; d < DIM; ++d) {
        const float df = pn[(cs * KMAX + k) * DIM + d] - cfin[(slot * KMAX + k) * DIM + d];
        l += df * df;
      }
    al += valid * (l / (float)(KMAX * DIM));
    ac += valid;
  }
  scal[0] = al / fmaxf(ac, 1.f);
}

__global__ void k_compact(const float* __restrict__ pn, float* __restrict__ scal) {
  if (threadIdx.x != 0) return;
  float total = 0.f;
  for (int c = 1; c < NUM_CLASSES; ++c) {
    float mv = 0.f;
    for (int d = 0; d < DIM; ++d) {
      float mean = 0.f;
      for (int k = 0; k < KMAX; ++k) mean += pn[(c * KMAX + k) * DIM + d];
      mean /= (float)KMAX;
      float var = 0.f;
      for (int k = 0; k < KMAX; ++k) {
        const float df = pn[(c * KMAX + k) * DIM + d] - mean;
        var += df * df;
      }
      mv += var / (float)(KMAX - 1);         // ddof=1
    }
    total += mv / (float)DIM;
  }
  scal[1] = total;
}

// ---------------------------------------------------------------------------
// Triplet loss: A-matrix = 25 prototypes (2 tiles), B-matrix = 16 anchors.
// Anchors sampled deterministically (stride 512), semi-hard negative picked by
// hash-argmax (substitute for gumbel).  All vectors unit-norm -> d = sqrt(2-2s).
// Tiny kernel (128 waves) so the full-sim exchange cost is irrelevant.
// ---------------------------------------------------------------------------
__global__ void __launch_bounds__(256) k_triplet(const _Float16* __restrict__ A,
                                                 const _Float16* __restrict__ protoA,
                                                 const int* __restrict__ pseudo,
                                                 float* __restrict__ tri) {
  const int lane = threadIdx.x & 31;
  const int wave = (blockIdx.x * blockDim.x + threadIdx.x) >> 5;   // 0..127
  const int m = lane & 15;
  const int aidx = wave * 16 + m;
  const int n = aidx * ANCH_STRIDE + (ANCH_STRIDE >> 1);

  const v16h a0 = load_row_tile(lane, protoA + m * DIM);
  const v16h a1 = load_row_tile(lane, protoA + (16 + m) * DIM);
  const v16h b  = load_row_tile(lane, A + (size_t)n * DIM);

  v8f d1 = {}, d2 = {};
  d1 = __builtin_amdgcn_wmma_f32_16x16x32_f16(false, a0, false, b, (short)0, d1, false, false);
  d2 = __builtin_amdgcn_wmma_f32_16x16x32_f16(false, a1, false, b, (short)0, d2, false, false);

  float s[CK];
#pragma unroll
  for (int j = 0; j < 8; ++j) {
    const float o = __shfl_xor(d1[j], 16, 32);
    s[j]     = (lane < 16) ? d1[j] : o;
    s[8 + j] = (lane < 16) ? o : d1[j];
  }
#pragma unroll
  for (int j = 0; j < 8; ++j) {
    const float o = __shfl_xor(d2[j], 16, 32);
    s[16 + j] = (lane < 16) ? d2[j] : o;     // protos 16..23
  }
  {
    const float o = __shfl_xor(d2[0], 16, 32);
    s[24] = (lane < 16) ? o : d2[0];         // proto 24 lives on the hi half (M=8)
  }

  const int p = pseudo[n];
  float w = (p > 0) ? 1.f : 0.f;
  int label = p < 0 ? 0 : (p > NUM_CLASSES - 1 ? NUM_CLASSES - 1 : p);

  float bests = -1e30f; int pidx = label * KMAX;
#pragma unroll
  for (int k = 0; k < KMAX; ++k) {           // argmax sim == argmin dist (own)
    const int j = label * KMAX + k;
    if (s[j] > bests) { bests = s[j]; pidx = j; }
  }
  const float spos = s[pidx];
  const float posd = sqrtf(fmaxf(2.f - 2.f * spos, 0.f));

  unsigned h1b = 0, h2b = 0; int semi = -1, rnd = -1;
#pragma unroll
  for (int j = 0; j < CK; ++j) {
    if (j / KMAX == label) continue;
    const float dj = sqrtf(fmaxf(2.f - 2.f * s[j], 0.f));
    const unsigned h2 = hashu((unsigned)(aidx * 64 + j) + 0x85ebca6bu);
    if (rnd < 0 || h2 > h2b) { h2b = h2; rnd = j; }
    if (dj > posd && dj < posd + MARGINF) {
      const unsigned h1 = hashu((unsigned)(aidx * 64 + j) + 0x9e3779b9u);
      if (semi < 0 || h1 > h1b) { h1b = h1; semi = j; }
    }
  }
  const int nidx = (semi >= 0) ? semi : rnd;
  const float dpos = sqrtf(fmaxf(2.f - 2.f * spos, 0.f) + EPSF);
  const float dneg = sqrtf(fmaxf(2.f - 2.f * s[nidx], 0.f) + EPSF);
  const float x = dpos - dneg + MARGINF;
  const float sp = (x > 20.f) ? x : log1pf(expf(x));
  float term = sp * w;
  if (lane >= 16) { term = 0.f; w = 0.f; }   // anchor duplicated on lane pair
#pragma unroll
  for (int off = 16; off; off >>= 1) {
    term += __shfl_xor(term, off, 32);
    w    += __shfl_xor(w, off, 32);
  }
  if (lane == 0) { tri[wave * 2] = term; tri[wave * 2 + 1] = w; }
}

__global__ void k_combine(const float* __restrict__ tri, const float* __restrict__ scal,
                          float* __restrict__ out) {
  if (threadIdx.x != 0) return;
  float st = 0.f, sw = 0.f;
  for (int i = 0; i < TRI_WAVES; ++i) { st += tri[2 * i]; sw += tri[2 * i + 1]; }
  const float proto_loss = st / fmaxf(sw, 1.f);
  out[0] = proto_loss + 1.0f * scal[0] + 0.1f * scal[1];   // ANCHOR_STRENGTH, LAMBDA_COMPACT
}

// ---------------------------------------------------------------------------
extern "C" void kernel_launch(void* const* d_in, const int* in_sizes, int n_in,
                              void* d_out, int out_size, void* d_ws, size_t ws_size,
                              hipStream_t stream) {
  (void)in_sizes; (void)n_in; (void)out_size; (void)ws_size;
  const float* feats  = (const float*)d_in[0];
  const float* protos = (const float*)d_in[1];
  const int*   pseudo = (const int*)d_in[2];
  const int*   gt     = (const int*)d_in[3];
  const int*   lab    = (const int*)d_in[4];

  char* ws = (char*)d_ws;
  _Float16* A      = (_Float16*)(ws + OFF_A);
  float*    part   = (float*)(ws + OFF_PART);
  float*    sums   = (float*)(ws + OFF_SUMS);
  float*    cent   = (float*)(ws + OFF_CENT);
  float*    cc2    = (float*)(ws + OFF_CC2);
  _Float16* centA  = (_Float16*)(ws + OFF_CENTA);
  float*    initpt = (float*)(ws + OFF_INITPT);
  float*    pn     = (float*)(ws + OFF_PROTO);
  _Float16* pa     = (_Float16*)(ws + OFF_PROTOA);
  float*    cls    = (float*)(ws + OFF_CLS);
  float*    cfin   = (float*)(ws + OFF_CFIN);
  float*    tri    = (float*)(ws + OFF_TRI);
  float*    scal   = (float*)(ws + OFF_SCAL);

  k_normalize<<<NVOX / 256, 256, 0, stream>>>(feats, A);
  k_protopack<<<1, 32, 0, stream>>>(protos, pn, pa);
  k_initcent<<<1, 32, 0, stream>>>(gt, lab, A, cent, initpt, cc2, centA);

  for (int it = 0; it < KM_ITERS; ++it) {
    k_assign<<<ASSIGN_WGS, ASSIGN_THREADS, 0, stream>>>(A, centA, cc2, gt, lab, part);
    k_reduce<<<1, 704, 0, stream>>>(part, sums);
    k_update<<<1, 640, 0, stream>>>(sums, initpt, cent, cc2, centA, cls, it);
  }

  k_finalize<<<1, 32, 0, stream>>>(cent, cls, cfin);
  k_anchor<<<1, 32, 0, stream>>>(pn, cfin, cls, lab, scal);
  k_compact<<<1, 32, 0, stream>>>(pn, scal);
  k_triplet<<<TRI_WAVES / 8, 256, 0, stream>>>(A, pa, pseudo, tri);
  k_combine<<<1, 32, 0, stream>>>(tri, scal, (float*)d_out);
}
